// NMR3DNet_40295383171095
// MI455X (gfx1250) — matrix operand order repaired and checked by reading, not compiled, not run on hardware
//
#include <hip/hip_runtime.h>
#include <cstddef>
#include <cstdint>

// ---------------------------------------------------------------------------
// NMR3DNet on MI455X (gfx1250). bf16 WMMA for all dense layers, f32 accum.
// Async global->LDS (CDNA5 TDM-class path) for the We stream in NNConv.
// Requires ~800 MB of workspace (dominated by the bf16 We tensor, 537 MB).
// ---------------------------------------------------------------------------

constexpr int NNODE  = 4096;
constexpr int NEDGE  = 16384;      // NNODE * DEG
constexpr int DEG    = 4;
constexpr int NHD    = 128;
constexpr int MED    = 256;
constexpr int EHD    = 256;
constexpr int NRBFD  = 32;
constexpr int KANG   = 4;
constexpr int NSTEPS = 5;
constexpr int NINTS  = 4;
constexpr int S2SIT  = 3;
constexpr int RHD    = 512;
constexpr int NING   = 74;
constexpr int EING   = 9;
constexpr int NGRAPH = 128;
constexpr int DAGG   = NHD * (1 + NSTEPS);   // 768

typedef __attribute__((ext_vector_type(16))) __bf16 v16bf;
typedef __attribute__((ext_vector_type(8)))  __bf16 v8bf;   // 16 B, align 16
typedef __attribute__((ext_vector_type(8)))  float  v8f;

__device__ inline __bf16 f2bf(float f) { return (__bf16)f; }  // native cvt, RTNE
__device__ inline float bfu_lo(unsigned u) {   // bf16 in bits [15:0]
  union { unsigned x; float f; } v; v.x = u << 16; return v.f;
}
__device__ inline float bfu_hi(unsigned u) {   // bf16 in bits [31:16]
  union { unsigned x; float f; } v; v.x = u & 0xFFFF0000u; return v.f;
}
__device__ inline float sigf(float x) { return 1.0f / (1.0f + expf(-x)); }

// ---------------------------------------------------------------------------
// Generic bf16 WMMA GEMM: C = act(A[f32 MxK] @ W[bf16 KxN] + bias + residual)
// 64x64 block tile, 32-deep K tile, 4 waves x 32 lanes (wave32).
// LDS staged in fragment order: A row-major (16B-aligned rows), B transposed,
// so every WMMA operand is fed by two ds_load_b128.
// act: 0=none 1=relu 2=tanh 3=silu 4=sigmoid 5=prelu(*alpha_ptr)
// ---------------------------------------------------------------------------
constexpr int GTM = 64, GTN = 64, GTK = 32;
constexpr int LDP = 40;   // padded stride: 80 B per row -> 16 B aligned

__global__ __launch_bounds__(128) void k_gemm(
    const float* __restrict__ A, int lda,
    const __bf16* __restrict__ W, int ldw,
    const float* __restrict__ bias,
    const float* __restrict__ resid, int ldr,
    void* __restrict__ Cv, int ldc,
    int M, int K, int Ncols, int act, const float* alpha_ptr, int out_bf16)
{
  __shared__ __align__(16) __bf16 As[GTM][LDP];   // [row][k]
  __shared__ __align__(16) __bf16 Bx[GTN][LDP];   // [col][k]  (transposed tile)
  const int tid  = threadIdx.x;
  const int wave = tid >> 5;
  const int lane = tid & 31;
  const int hi   = lane >> 4;
  const int l15  = lane & 15;
  const int m0 = blockIdx.y * GTM;
  const int n0 = blockIdx.x * GTN;

  v8f acc[4];
  for (int t = 0; t < 4; ++t) acc[t] = v8f{0.f,0.f,0.f,0.f,0.f,0.f,0.f,0.f};

  const bool w_even = ((ldw & 1) == 0);

  for (int kk = 0; kk < K; kk += GTK) {
    // ---- A tile: f32 -> bf16, packed dword stores ----
    for (int i = tid; i < GTM * (GTK / 2); i += 128) {
      int r = i >> 4, c2 = (i & 15) * 2;
      int gr = m0 + r, gc = kk + c2;
      float f0 = 0.f, f1 = 0.f;
      if (gr < M) {
        if (gc     < K) f0 = A[(size_t)gr * lda + gc];
        if (gc + 1 < K) f1 = A[(size_t)gr * lda + gc + 1];
      }
      union { unsigned u; __bf16 b[2]; } pk;
      pk.b[0] = f2bf(f0); pk.b[1] = f2bf(f1);
      *(unsigned*)&As[r][c2] = pk.u;
    }
    // ---- B tile: transposed store (Bx[col][k]) ----
    if (w_even) {
      for (int i = tid; i < GTK * (GTN / 2); i += 128) {
        int r = i >> 5, c2 = (i & 31) * 2;
        int gr = kk + r, gc = n0 + c2;
        union { unsigned u; __bf16 b[2]; } pk; pk.u = 0;
        if (gr < K && gc < Ncols)
          pk.u = *(const unsigned*)(W + (size_t)gr * ldw + gc);   // ldw,N even
        Bx[c2][r]     = pk.b[0];
        Bx[c2 + 1][r] = pk.b[1];
      }
    } else {
      for (int i = tid; i < GTK * GTN; i += 128) {
        int r = i >> 6, c = i & 63;
        int gr = kk + r, gc = n0 + c;
        Bx[c][r] = (gr < K && gc < Ncols) ? W[(size_t)gr * ldw + gc] : f2bf(0.f);
      }
    }
    __syncthreads();

    // A fragment: lane row = wave*16+l15, K halves per documented layout;
    // two contiguous 16 B runs -> 2x ds_load_b128.
    const int arow = wave * 16 + l15;
    const int akb  = hi * 8;
    v8bf alo = *(const v8bf*)&As[arow][akb];
    v8bf ahi = *(const v8bf*)&As[arow][16 + akb];
    v16bf af = __builtin_shufflevector(alo, ahi,
        0,1,2,3,4,5,6,7,8,9,10,11,12,13,14,15);

    #pragma unroll
    for (int t = 0; t < 4; ++t) {
      const int col = t * 16 + l15;
      const int bkb = hi * 8;
      v8bf blo = *(const v8bf*)&Bx[col][bkb];
      v8bf bhi = *(const v8bf*)&Bx[col][bkb + 16];
      v16bf bf = __builtin_shufflevector(blo, bhi,
          0,1,2,3,4,5,6,7,8,9,10,11,12,13,14,15);
      acc[t] = __builtin_amdgcn_wmma_f32_16x16x32_bf16(
          false, af, false, bf, (short)0, acc[t], false, false);
    }
    __syncthreads();
  }

  const float alpha = alpha_ptr ? *alpha_ptr : 0.f;
  for (int t = 0; t < 4; ++t) {
    int gn = n0 + t * 16 + l15;
    #pragma unroll
    for (int r = 0; r < 8; ++r) {
      int gm = m0 + wave * 16 + hi * 8 + r;
      if (gm < M && gn < Ncols) {
        float v = acc[t][r];
        if (bias)  v += bias[gn];
        if (resid) v += resid[(size_t)gm * ldr + gn];
        switch (act) {
          case 1: v = fmaxf(v, 0.f); break;
          case 2: v = tanhf(v); break;
          case 3: v = v * sigf(v); break;
          case 4: v = sigf(v); break;
          case 5: v = (v >= 0.f) ? v : alpha * v; break;
          default: break;
        }
        if (out_bf16) ((__bf16*)Cv)[(size_t)gm * ldc + gn] = f2bf(v);
        else          ((float*) Cv)[(size_t)gm * ldc + gn] = v;
      }
    }
  }
}

// ---- weight conversion -----------------------------------------------------
__global__ void k_f2bf(const float* __restrict__ s, __bf16* __restrict__ d, int n) {
  int i = blockIdx.x * blockDim.x + threadIdx.x;
  if (i < n) d[i] = f2bf(s[i]);
}
__global__ void k_f2bfT(const float* __restrict__ s, __bf16* __restrict__ d, int R, int C) {
  int i = blockIdx.x * blockDim.x + threadIdx.x;
  if (i >= R * C) return;
  int r = i / C, c = i % C;
  d[(size_t)c * R + r] = f2bf(s[(size_t)r * C + c]);
}

// ---- NNConv message + static segment-sum (dst = repeat(arange(N),4)) -------
// 2 waves per node. Each 32 KB edge matrix is staged into LDS with
// GLOBAL_LOAD_ASYNC_TO_LDS_B128 (ASYNCcnt), double-buffered so the async
// engine streams matrix j+1 while the waves accumulate matrix j from LDS.
__global__ __launch_bounds__(64) void k_nnconv(
    const float* __restrict__ h, const __bf16* __restrict__ We,
    const int* __restrict__ src, float* __restrict__ agg)
{
  const int n = blockIdx.x;
  const int t = threadIdx.x;            // 0..63
  __shared__ __align__(16) float  hsm[DEG][NHD];
  __shared__ __align__(16) __bf16 wbuf[2][NHD * NHD];   // 2 x 32 KB

  // preload the four h[src] rows
  for (int i = t; i < DEG * NHD; i += 64) {
    int j = i >> 7, c = i & (NHD - 1);
    hsm[j][c] = h[(size_t)src[n * DEG + j] * NHD + c];
  }

  // 32 async b128 issues per wave per matrix (64 lanes x 16 B x 32 iters = 32 KB)
  auto issue = [&](int buf, int e) {
    const char* g = (const char*)(We + (size_t)e * (NHD * NHD));
    unsigned lb = (unsigned)(uintptr_t)&wbuf[buf][0];
    #pragma unroll 4
    for (int it = 0; it < (NHD * NHD * 2) / (64 * 16); ++it) {
      unsigned lo = lb + (unsigned)((it * 64 + t) * 16);
      unsigned long long ga = (unsigned long long)(g + (size_t)(it * 64 + t) * 16);
      asm volatile("global_load_async_to_lds_b128 %0, %1, off"
                   :: "v"(lo), "v"(ga) : "memory");
    }
  };

  issue(0, n * DEG);
  float a0 = 0.f, a1 = 0.f;
  for (int j = 0; j < DEG; ++j) {
    const int cur = j & 1;
    if (j + 1 < DEG) {
      issue(1 - cur, n * DEG + j + 1);
      asm volatile("s_wait_asynccnt 32" ::: "memory");   // batch j complete (in-order)
    } else {
      asm volatile("s_wait_asynccnt 0" ::: "memory");
    }
    __syncthreads();
    const unsigned* wb = (const unsigned*)&wbuf[cur][0];  // dims (2t, 2t+1)
    float hcj;
    #pragma unroll 4
    for (int c = 0; c < NHD; ++c) {
      unsigned p = wb[c * 64 + t];
      hcj = hsm[j][c];
      a0 += hcj * bfu_lo(p);
      a1 += hcj * bfu_hi(p);
    }
    __syncthreads();   // all reads of wbuf[cur] done before it is refilled
  }
  *(float2*)(agg + (size_t)n * NHD + 2 * t) = make_float2(a0, a1);
}

// ---- GRU / LSTM elementwise ------------------------------------------------
__global__ void k_gru(const float* __restrict__ gi, const float* __restrict__ gh,
                      const float* __restrict__ h, float* __restrict__ hout) {
  int i = blockIdx.x * blockDim.x + threadIdx.x;
  if (i >= NNODE * NHD) return;
  int n = i / NHD, d = i % NHD;
  const float* a = gi + (size_t)n * 3 * NHD;
  const float* b = gh + (size_t)n * 3 * NHD;
  float r  = sigf(a[d] + b[d]);
  float z  = sigf(a[NHD + d] + b[NHD + d]);
  float nn = tanhf(a[2*NHD + d] + r * b[2*NHD + d]);
  hout[i] = (1.f - z) * nn + z * h[i];
}
__global__ void k_lstm(const float* __restrict__ gates, float* __restrict__ cl,
                       float* __restrict__ hl) {
  int i = blockIdx.x * blockDim.x + threadIdx.x;
  if (i >= NGRAPH * DAGG) return;
  int b = i / DAGG, d = i % DAGG;
  const float* g = gates + (size_t)b * 4 * DAGG;
  float c = sigf(g[DAGG + d]) * cl[i] + sigf(g[d]) * tanhf(g[2*DAGG + d]);
  cl[i] = c;
  hl[i] = sigf(g[3*DAGG + d]) * tanhf(c);
}

// ---- 3D geometry -----------------------------------------------------------
__global__ void k_edge3d(const float* __restrict__ pos, const int* __restrict__ src,
                         const int* __restrict__ dst, float* __restrict__ ef3,
                         float* __restrict__ d1) {
  int e = blockIdx.x * blockDim.x + threadIdx.x;
  if (e >= NEDGE) return;
  int s = src[e], d = dst[e];
  float dx = pos[s*3+0] - pos[d*3+0];
  float dy = pos[s*3+1] - pos[d*3+1];
  float dz = pos[s*3+2] - pos[d*3+2];
  float dist = sqrtf(dx*dx + dy*dy + dz*dz);
  d1[e] = dist;
  const float step  = 8.0f / 31.0f;
  const float coeff = -0.5f / (step * step);
  float* o = ef3 + (size_t)e * (EHD + NRBFD) + EHD;
  for (int k = 0; k < NRBFD; ++k) {
    float t = dist - step * (float)k;
    o[k] = expf(coeff * t * t);
  }
}

__global__ void k_angle(const float* __restrict__ pos, const int* __restrict__ src,
                        float* __restrict__ ang) {
  int n = blockIdx.x * blockDim.x + threadIdx.x;
  if (n >= NNODE) return;
  float u[DEG][3];
  for (int j = 0; j < DEG; ++j) {
    int s = src[n * DEG + j];
    float vx = pos[s*3+0] - pos[n*3+0];
    float vy = pos[s*3+1] - pos[n*3+1];
    float vz = pos[s*3+2] - pos[n*3+2];
    float nm = fmaxf(sqrtf(vx*vx + vy*vy + vz*vz), 1e-8f);
    u[j][0] = vx / nm; u[j][1] = vy / nm; u[j][2] = vz / nm;
  }
  float s[2 * KANG + 2];
  for (int t = 0; t < 2 * KANG + 2; ++t) s[t] = 0.f;
  for (int a = 0; a < DEG; ++a)
    for (int b = a + 1; b < DEG; ++b) {
      float c = u[a][0]*u[b][0] + u[a][1]*u[b][1] + u[a][2]*u[b][2];
      c = fminf(1.f, fmaxf(-1.f, c));
      float th = acosf(c);
      s[0] += cosf(th); s[1] += sinf(th);
      for (int k = 1; k <= KANG; ++k) {
        s[2*k]     += cosf((float)k * th);
        s[2*k + 1] += sinf((float)k * th);
      }
    }
  const float inv = 1.f / 6.f;   // 6 valid pairs per node (deg==4)
  for (int t = 0; t < 2 * KANG + 2; ++t) ang[(size_t)n * (2*KANG+2) + t] = s[t] * inv;
}

// ---- SchNet / EGNN edge aggregation (sorted dst, degree 4) -----------------
__global__ void k_sch_agg(const float* __restrict__ xs, const float* __restrict__ ew,
                          const int* __restrict__ src, float* __restrict__ o) {
  int n = blockIdx.x, d = threadIdx.x;
  float acc = 0.f;
  for (int j = 0; j < DEG; ++j) {
    int e = n * DEG + j;
    acc += xs[(size_t)src[e] * NHD + d] * ew[(size_t)e * NHD + d];
  }
  o[(size_t)n * NHD + d] = acc;
}
__global__ void k_edge_sum(const float* __restrict__ em, float* __restrict__ nm, int cols) {
  int n = blockIdx.x;
  for (int d = threadIdx.x; d < cols; d += blockDim.x) {
    float acc = 0.f;
    for (int j = 0; j < DEG; ++j) acc += em[(size_t)(n * DEG + j) * cols + d];
    nm[(size_t)n * cols + d] = acc;
  }
}
__global__ void k_ei(const float* __restrict__ xe, const float* __restrict__ ef3,
                     const float* __restrict__ d1, const int* __restrict__ src,
                     const int* __restrict__ dst, float* __restrict__ ei) {
  constexpr int CI = 2 * NHD + (EHD + NRBFD) + 1;   // 545
  int e = blockIdx.x;
  int s = src[e], dd = dst[e];
  for (int c = threadIdx.x; c < CI; c += blockDim.x) {
    float v;
    if      (c < NHD)        v = xe[(size_t)s  * NHD + c];
    else if (c < 2*NHD)      v = xe[(size_t)dd * NHD + (c - NHD)];
    else if (c < CI - 1)     v = ef3[(size_t)e * (EHD + NRBFD) + (c - 2*NHD)];
    else                     v = d1[e];
    ei[(size_t)e * CI + c] = v;
  }
}

// ---- small elementwise helpers --------------------------------------------
__global__ void k_fill(float* p, float v, int n) {
  int i = blockIdx.x * blockDim.x + threadIdx.x;
  if (i < n) p[i] = v;
}
__global__ void k_add(float* __restrict__ a, const float* __restrict__ b, int n) {
  int i = blockIdx.x * blockDim.x + threadIdx.x;
  if (i < n) a[i] += b[i];
}
__global__ void k_mix(const float* __restrict__ g, const float* __restrict__ a,
                      const float* __restrict__ b, float* __restrict__ o, int n) {
  int i = blockIdx.x * blockDim.x + threadIdx.x;
  if (i < n) o[i] = g[i] * a[i] + (1.f - g[i]) * b[i];
}
__global__ void k_copy_cols(const float* __restrict__ s, int lds_, float* __restrict__ d,
                            int ldd, int rows, int cols, int doff) {
  int i = blockIdx.x * blockDim.x + threadIdx.x;
  if (i >= rows * cols) return;
  int r = i / cols, c = i % cols;
  d[(size_t)r * ldd + doff + c] = s[(size_t)r * lds_ + c];
}
__global__ void k_bcast(const float* __restrict__ G, const int* __restrict__ batch,
                        float* __restrict__ d, int ldd, int doff, int rows, int cols) {
  int i = blockIdx.x * blockDim.x + threadIdx.x;
  if (i >= rows * cols) return;
  int r = i / cols, c = i % cols;
  d[(size_t)r * ldd + doff + c] = G[(size_t)batch[r] * cols + c];
}
__global__ void k_mask(const float* __restrict__ y, const unsigned char* __restrict__ m,
                       float* __restrict__ out, int n) {
  int i = blockIdx.x * blockDim.x + threadIdx.x;
  if (i < n) out[i] = m[i] ? y[i] : 0.f;
}

// ---- set2set attention -----------------------------------------------------
__global__ void k_attn_dot(const float* __restrict__ Hn, const float* __restrict__ q,
                           const int* __restrict__ batch, float* __restrict__ e_) {
  int wave = threadIdx.x >> 5, lane = threadIdx.x & 31;
  int n = blockIdx.x * (blockDim.x >> 5) + wave;
  if (n >= NNODE) return;
  const float* hr = Hn + (size_t)n * DAGG;
  const float* qr = q  + (size_t)batch[n] * DAGG;
  float v = 0.f;
  for (int c = lane; c < DAGG; c += 32) v += hr[c] * qr[c];
  for (int off = 16; off; off >>= 1) v += __shfl_down(v, off);
  if (lane == 0) e_[n] = v;
}
__global__ void k_attn_soft(const float* __restrict__ e_, float* __restrict__ a_) {
  int g = blockIdx.x, lane = threadIdx.x;         // 32 nodes per graph
  float v = e_[g * 32 + lane];
  float m = v;
  for (int off = 16; off; off >>= 1) m = fmaxf(m, __shfl_xor(m, off));
  float ex = expf(v - m);
  float s = ex;
  for (int off = 16; off; off >>= 1) s += __shfl_xor(s, off);
  a_[g * 32 + lane] = ex / s;
}
__global__ void k_attn_r(const float* __restrict__ a_, const float* __restrict__ Hn,
                         float* __restrict__ r_) {
  int g = blockIdx.x;
  for (int d = threadIdx.x; d < DAGG; d += blockDim.x) {
    float acc = 0.f;
    for (int j = 0; j < 32; ++j)
      acc += a_[g * 32 + j] * Hn[(size_t)(g * 32 + j) * DAGG + d];
    r_[(size_t)g * DAGG + d] = acc;
  }
}

// ---------------------------------------------------------------------------
extern "C" void kernel_launch(void* const* d_in, const int* in_sizes, int n_in,
                              void* d_out, int out_size, void* d_ws, size_t ws_size,
                              hipStream_t stream) {
  (void)in_sizes; (void)n_in; (void)out_size; (void)ws_size;

  // ---- inputs (setup_inputs dict order) ----
  const float* x         = (const float*)d_in[0];
  const int*   eidx      = (const int*)  d_in[1];
  const float* edge_attr = (const float*)d_in[2];
  const float* pos       = (const float*)d_in[3];
  const int*   batch     = (const int*)  d_in[4];
  const unsigned char* mask = (const unsigned char*)d_in[5];
  const int* src = eidx;
  const int* dst = eidx + NEDGE;

  // ---- params (jax pytree: dict keys sorted; per-linear 'b' before 'w') ----
  int pi = 6;
  auto nxt = [&]() -> const float* { return (const float*)d_in[pi++]; };
  const float *cm_b0 = nxt(), *cm_w0 = nxt(), *cm_b1 = nxt(), *cm_w1 = nxt();   // comenet_proj
  const float *e2b[4], *e2w[4];                                                  // edge_mlp_2d
  for (int i = 0; i < 4; ++i) { e2b[i] = nxt(); e2w[i] = nxt(); }
  const float *ep_b = nxt(), *ep_w = nxt();                                      // edge_proj_3d
  struct MLP4 { const float *eb0,*ew0,*eb1,*ew1,*nb0,*nw0,*nb1,*nw1; };
  MLP4 eg[4];                                                                    // egnn
  for (int i = 0; i < 4; ++i) eg[i] = { nxt(),nxt(),nxt(),nxt(),nxt(),nxt(),nxt(),nxt() };
  const float *f23_b = nxt(), *f23_w = nxt();                                    // fusion_gate_2d3d
  const float *f3_b  = nxt(), *f3_w  = nxt();                                    // fusion_gate_3d
  const float *g_bhh = nxt(), *g_bih = nxt(), *g_whh = nxt(), *g_wih = nxt();    // gru
  const float *rt_b  = nxt(), *rt_w  = nxt();                                    // nnconv_root
  const float *npb[4], *npw[4];                                                  // node_proj
  for (int i = 0; i < 4; ++i) { npb[i] = nxt(); npw[i] = nxt(); }
  const float *pl0b = nxt(), *pl0w = nxt();                                      // predictor.l0
  const float *pl1b = nxt(), *pl1w = nxt();                                      // predictor.l1
  const float *pl2b = nxt(), *pl2w = nxt();                                      // predictor.l2
  const float *pp0  = nxt(), *pp1  = nxt();                                      // p0, p1 scalars
  MLP4 sc[4];                                                                    // schnet
  for (int i = 0; i < 4; ++i) sc[i] = { nxt(),nxt(),nxt(),nxt(),nxt(),nxt(),nxt(),nxt() };
  const float *s_bhh = nxt(), *s_bih = nxt(), *s_whh = nxt(), *s_wih = nxt();    // set2set_lstm

  // ---- deterministic workspace bump allocator ----
  char*  wsb = (char*)d_ws;
  size_t cur = 0;
  auto alloc = [&](size_t bytes) -> char* {
    char* p = wsb + cur;
    cur = (cur + bytes + 255) & ~(size_t)255;
    return p;
  };
  auto cvt = [&](const float* w, int n) -> const __bf16* {
    __bf16* p = (__bf16*)alloc((size_t)n * 2);
    k_f2bf<<<dim3((n + 255) / 256), 256, 0, stream>>>(w, p, n);
    return p;
  };
  auto cvtT = [&](const float* w, int R, int C) -> const __bf16* {
    __bf16* p = (__bf16*)alloc((size_t)R * C * 2);
    int n = R * C;
    k_f2bfT<<<dim3((n + 255) / 256), 256, 0, stream>>>(w, p, R, C);
    return p;
  };
  auto gemm = [&](const float* A, int lda, const __bf16* W, int ldw, const float* bias,
                  const float* resid, int ldr, void* C, int ldc,
                  int M, int K, int N, int act, const float* aptr, int obf) {
    dim3 g((N + GTN - 1) / GTN, (M + GTM - 1) / GTM);
    k_gemm<<<g, 128, 0, stream>>>(A, lda, W, ldw, bias, resid, ldr, C, ldc,
                                  M, K, N, act, aptr, obf);
  };

  // ---- bf16 weights ----
  const __bf16* b_np[4];  for (int i=0;i<4;++i) b_np[i] = cvt(npw[i], (i==0?NING:MED) * (i==3?NHD:MED));
  const __bf16* b_e2[4];  for (int i=0;i<4;++i) b_e2[i] = cvt(e2w[i], (i==0?EING:MED) * (i==3?NHD*NHD:MED));
  const __bf16* b_rt  = cvt(rt_w, NHD * NHD);
  const __bf16* b_gih = cvtT(g_wih, 3 * NHD, NHD);   // (NHD x 3NHD)
  const __bf16* b_ghh = cvtT(g_whh, 3 * NHD, NHD);
  const __bf16* b_ep  = cvt(ep_w, EING * EHD);
  const __bf16* b_cm0 = cvt(cm_w0, (2*KANG+2) * NHD);
  const __bf16* b_cm1 = cvt(cm_w1, NHD * NHD);
  const __bf16 *b_se0[4], *b_se1[4], *b_sn0[4], *b_sn1[4];
  for (int i = 0; i < 4; ++i) {
    b_se0[i] = cvt(sc[i].ew0, (EHD + NRBFD) * EHD);
    b_se1[i] = cvt(sc[i].ew1, EHD * NHD);
    b_sn0[i] = cvt(sc[i].nw0, NHD * EHD);
    b_sn1[i] = cvt(sc[i].nw1, EHD * NHD);
  }
  constexpr int CEI = 2 * NHD + EHD + NRBFD + 1;   // 545
  const __bf16 *b_ee0[4], *b_ee1[4], *b_en0[4], *b_en1[4];
  for (int i = 0; i < 4; ++i) {
    b_ee0[i] = cvt(eg[i].ew0, CEI * EHD);
    b_ee1[i] = cvt(eg[i].ew1, EHD * EHD);
    b_en0[i] = cvt(eg[i].nw0, (NHD + EHD) * EHD);
    b_en1[i] = cvt(eg[i].nw1, EHD * NHD);
  }
  const __bf16* b_f3  = cvt(f3_w,  2 * NHD * NHD);
  const __bf16* b_f23 = cvt(f23_w, 2 * NHD * NHD);
  const __bf16* b_sih = cvtT(s_wih, 4 * DAGG, 2 * DAGG);   // (1536 x 3072)
  const __bf16* b_shh = cvtT(s_whh, 4 * DAGG, DAGG);       // (768  x 3072)
  const __bf16* b_pl0 = cvt(pl0w, 3 * DAGG * RHD);
  const __bf16* b_pl1 = cvt(pl1w, RHD * RHD);
  const __bf16* b_pl2 = cvt(pl2w, RHD * 1);

  // ---- activation buffers ----
  __bf16* We   = (__bf16*)alloc((size_t)NEDGE * NHD * NHD * 2);        // 537 MB
  float*  tE0  = (float*)alloc((size_t)NEDGE * MED * 4);
  float*  tE1  = (float*)alloc((size_t)NEDGE * MED * 4);
  float*  eiB  = (float*)alloc((size_t)NEDGE * CEI * 4);
  float*  ef3  = (float*)alloc((size_t)NEDGE * (EHD + NRBFD) * 4);
  float*  ew   = (float*)alloc((size_t)NEDGE * NHD * 4);
  float*  d1   = (float*)alloc((size_t)NEDGE * 4);
  float*  hs   = (float*)alloc((size_t)(NSTEPS + 1) * NNODE * NHD * 4);
  float*  agg  = (float*)alloc((size_t)NNODE * NHD * 4);
  float*  mbuf = (float*)alloc((size_t)NNODE * NHD * 4);
  float*  gi   = (float*)alloc((size_t)NNODE * 3 * NHD * 4);
  float*  gh   = (float*)alloc((size_t)NNODE * 3 * NHD * 4);
  float*  xs   = (float*)alloc((size_t)NNODE * NHD * 4);
  float*  xe   = (float*)alloc((size_t)NNODE * NHD * 4);
  float*  x3d  = (float*)alloc((size_t)NNODE * NHD * 4);
  float*  hf   = (float*)alloc((size_t)NNODE * NHD * 4);
  float*  gbuf = (float*)alloc((size_t)NNODE * NHD * 4);
  float*  angb = (float*)alloc((size_t)NNODE * (2*KANG+2) * 4);
  float*  aemb = (float*)alloc((size_t)NNODE * NHD * 4);
  float*  tN0  = (float*)alloc((size_t)NNODE * (NHD + EHD) * 4);
  float*  tN1  = (float*)alloc((size_t)NNODE * EHD * 4);
  float*  Hn   = (float*)alloc((size_t)NNODE * DAGG * 4);
  float*  eatt = (float*)alloc((size_t)NNODE * 4);
  float*  aatt = (float*)alloc((size_t)NNODE * 4);
  float*  qst  = (float*)alloc((size_t)NGRAPH * 2 * DAGG * 4);
  float*  hl   = (float*)alloc((size_t)NGRAPH * DAGG * 4);
  float*  cl   = (float*)alloc((size_t)NGRAPH * DAGG * 4);
  float*  gat  = (float*)alloc((size_t)NGRAPH * 4 * DAGG * 4);
  float*  rbuf = (float*)alloc((size_t)NGRAPH * DAGG * 4);
  float*  Pcat = (float*)alloc((size_t)NNODE * 3 * DAGG * 4);
  float*  z0   = (float*)alloc((size_t)NNODE * RHD * 4);
  float*  z1   = (float*)alloc((size_t)NNODE * RHD * 4);
  float*  ybuf = (float*)alloc((size_t)NNODE * 4);

  auto gN  = [&](int n) { return dim3((n + 255) / 256); };

  // ==== 1. node_proj: x -> h0 ====
  gemm(x,   NING, b_np[0], MED, npb[0], nullptr, 0, tN0, MED, NNODE, NING, MED, 1, nullptr, 0);
  gemm(tN0, MED,  b_np[1], MED, npb[1], nullptr, 0, tN1, MED, NNODE, MED,  MED, 1, nullptr, 0);
  gemm(tN1, MED,  b_np[2], MED, npb[2], nullptr, 0, tN0, MED, NNODE, MED,  MED, 1, nullptr, 0);
  gemm(tN0, MED,  b_np[3], NHD, npb[3], nullptr, 0, hs,  NHD, NNODE, MED,  NHD, 2, nullptr, 0);

  // ==== 2. edge_mlp_2d: edge_attr -> We (bf16, relu) ====
  gemm(edge_attr, EING, b_e2[0], MED, e2b[0], nullptr, 0, tE0, MED, NEDGE, EING, MED, 1, nullptr, 0);
  gemm(tE0, MED,  b_e2[1], MED, e2b[1], nullptr, 0, tE1, MED, NEDGE, MED, MED, 1, nullptr, 0);
  gemm(tE1, MED,  b_e2[2], MED, e2b[2], nullptr, 0, tE0, MED, NEDGE, MED, MED, 1, nullptr, 0);
  gemm(tE0, MED,  b_e2[3], NHD*NHD, e2b[3], nullptr, 0, We, NHD*NHD, NEDGE, MED, NHD*NHD, 1, nullptr, 1);

  // ==== 3. NNConv + GRU (5 steps) ====
  for (int s = 0; s < NSTEPS; ++s) {
    float* hcur = hs + (size_t)s * NNODE * NHD;
    float* hnew = hs + (size_t)(s + 1) * NNODE * NHD;
    k_nnconv<<<NNODE, 64, 0, stream>>>(hcur, We, src, agg);
    gemm(hcur, NHD, b_rt,  NHD,   rt_b,  agg, NHD, mbuf, NHD,   NNODE, NHD, NHD,   0, nullptr, 0);
    gemm(mbuf, NHD, b_gih, 3*NHD, g_bih, nullptr, 0, gi, 3*NHD, NNODE, NHD, 3*NHD, 0, nullptr, 0);
    gemm(hcur, NHD, b_ghh, 3*NHD, g_bhh, nullptr, 0, gh, 3*NHD, NNODE, NHD, 3*NHD, 0, nullptr, 0);
    k_gru<<<gN(NNODE * NHD), 256, 0, stream>>>(gi, gh, hcur, hnew);
  }
  float* h5 = hs + (size_t)NSTEPS * NNODE * NHD;

  // ==== 4. 3D edge features ====
  gemm(edge_attr, EING, b_ep, EHD, ep_b, nullptr, 0, ef3, EHD + NRBFD, NEDGE, EING, EHD, 0, nullptr, 0);
  k_edge3d<<<gN(NEDGE), 256, 0, stream>>>(pos, src, dst, ef3, d1);

  // ==== 5. angle features + comenet ====
  k_angle<<<gN(NNODE), 256, 0, stream>>>(pos, src, angb);
  gemm(angb, 2*KANG+2, b_cm0, NHD, cm_b0, nullptr, 0, tN1, NHD, NNODE, 2*KANG+2, NHD, 3, nullptr, 0);
  gemm(tN1,  NHD,      b_cm1, NHD, cm_b1, nullptr, 0, aemb, NHD, NNODE, NHD, NHD, 0, nullptr, 0);

  // ==== 6. SchNet branch ====
  k_copy_cols<<<gN(NNODE*NHD), 256, 0, stream>>>(h5, NHD, xs, NHD, NNODE, NHD, 0);
  for (int it = 0; it < NINTS; ++it) {
    gemm(ef3, EHD+NRBFD, b_se0[it], EHD, sc[it].eb0, nullptr, 0, tE0, EHD, NEDGE, EHD+NRBFD, EHD, 3, nullptr, 0);
    gemm(tE0, EHD,       b_se1[it], NHD, sc[it].eb1, nullptr, 0, ew,  NHD, NEDGE, EHD, NHD, 0, nullptr, 0);
    k_sch_agg<<<NNODE, NHD, 0, stream>>>(xs, ew, src, tN1);
    gemm(tN1, NHD, b_sn0[it], EHD, sc[it].nb0, nullptr, 0, tN0, EHD, NNODE, NHD, EHD, 3, nullptr, 0);
    gemm(tN0, EHD, b_sn1[it], NHD, sc[it].nb1, xs, NHD,  xs,  NHD, NNODE, EHD, NHD, 0, nullptr, 0);
  }
  k_add<<<gN(NNODE*NHD), 256, 0, stream>>>(xs, aemb, NNODE * NHD);

  // ==== 7. EGNN branch ====
  k_copy_cols<<<gN(NNODE*NHD), 256, 0, stream>>>(h5, NHD, xe, NHD, NNODE, NHD, 0);
  for (int it = 0; it < NINTS; ++it) {
    k_ei<<<NEDGE, 128, 0, stream>>>(xe, ef3, d1, src, dst, eiB);
    gemm(eiB, CEI, b_ee0[it], EHD, eg[it].eb0, nullptr, 0, tE0, EHD, NEDGE, CEI, EHD, 3, nullptr, 0);
    gemm(tE0, EHD, b_ee1[it], EHD, eg[it].eb1, nullptr, 0, tE1, EHD, NEDGE, EHD, EHD, 0, nullptr, 0);
    k_edge_sum<<<NNODE, 256, 0, stream>>>(tE1, tN1, EHD);
    k_copy_cols<<<gN(NNODE*NHD), 256, 0, stream>>>(xe,  NHD, tN0, NHD+EHD, NNODE, NHD, 0);
    k_copy_cols<<<gN(NNODE*EHD), 256, 0, stream>>>(tN1, EHD, tN0, NHD+EHD, NNODE, EHD, NHD);
    gemm(tN0, NHD+EHD, b_en0[it], EHD, eg[it].nb0, nullptr, 0, tN1, EHD, NNODE, NHD+EHD, EHD, 3, nullptr, 0);
    gemm(tN1, EHD,     b_en1[it], NHD, eg[it].nb1, xe, NHD,  xe,  NHD, NNODE, EHD, NHD, 0, nullptr, 0);
  }
  k_add<<<gN(NNODE*NHD), 256, 0, stream>>>(xe, aemb, NNODE * NHD);

  // ==== 8. fusion gates ====
  k_copy_cols<<<gN(NNODE*NHD), 256, 0, stream>>>(xs, NHD, tN0, 2*NHD, NNODE, NHD, 0);
  k_copy_cols<<<gN(NNODE*NHD), 256, 0, stream>>>(xe, NHD, tN0, 2*NHD, NNODE, NHD, NHD);
  gemm(tN0, 2*NHD, b_f3, NHD, f3_b, nullptr, 0, gbuf, NHD, NNODE, 2*NHD, NHD, 4, nullptr, 0);
  k_mix<<<gN(NNODE*NHD), 256, 0, stream>>>(gbuf, xs, xe, x3d, NNODE * NHD);
  k_copy_cols<<<gN(NNODE*NHD), 256, 0, stream>>>(h5,  NHD, tN0, 2*NHD, NNODE, NHD, 0);
  k_copy_cols<<<gN(NNODE*NHD), 256, 0, stream>>>(x3d, NHD, tN0, 2*NHD, NNODE, NHD, NHD);
  gemm(tN0, 2*NHD, b_f23, NHD, f23_b, nullptr, 0, gbuf, NHD, NNODE, 2*NHD, NHD, 4, nullptr, 0);
  k_mix<<<gN(NNODE*NHD), 256, 0, stream>>>(gbuf, h5, x3d, hf, NNODE * NHD);

  // ==== 9. Hn = concat(h0..h4, hf) ====
  for (int s = 0; s < NSTEPS; ++s)
    k_copy_cols<<<gN(NNODE*NHD), 256, 0, stream>>>(hs + (size_t)s * NNODE * NHD, NHD,
                                                   Hn, DAGG, NNODE, NHD, s * NHD);
  k_copy_cols<<<gN(NNODE*NHD), 256, 0, stream>>>(hf, NHD, Hn, DAGG, NNODE, NHD, NSTEPS * NHD);

  // ==== 10. set2set ====
  k_fill<<<gN(NGRAPH*2*DAGG), 256, 0, stream>>>(qst, 0.f, NGRAPH * 2 * DAGG);
  k_fill<<<gN(NGRAPH*DAGG),   256, 0, stream>>>(hl,  0.f, NGRAPH * DAGG);
  k_fill<<<gN(NGRAPH*DAGG),   256, 0, stream>>>(cl,  0.f, NGRAPH * DAGG);
  for (int it = 0; it < S2SIT; ++it) {
    gemm(qst, 2*DAGG, b_sih, 4*DAGG, s_bih, nullptr, 0, gat, 4*DAGG, NGRAPH, 2*DAGG, 4*DAGG, 0, nullptr, 0);
    gemm(hl,  DAGG,   b_shh, 4*DAGG, s_bhh, gat, 4*DAGG, gat, 4*DAGG, NGRAPH, DAGG,   4*DAGG, 0, nullptr, 0);
    k_lstm<<<gN(NGRAPH*DAGG), 256, 0, stream>>>(gat, cl, hl);
    k_attn_dot<<<dim3(NNODE / 8), 256, 0, stream>>>(Hn, hl, batch, eatt);
    k_attn_soft<<<NGRAPH, 32, 0, stream>>>(eatt, aatt);
    k_attn_r<<<NGRAPH, 256, 0, stream>>>(aatt, Hn, rbuf);
    k_copy_cols<<<gN(NGRAPH*DAGG), 256, 0, stream>>>(hl,   DAGG, qst, 2*DAGG, NGRAPH, DAGG, 0);
    k_copy_cols<<<gN(NGRAPH*DAGG), 256, 0, stream>>>(rbuf, DAGG, qst, 2*DAGG, NGRAPH, DAGG, DAGG);
  }

  // ==== 11. predictor ====
  k_copy_cols<<<gN(NNODE*DAGG), 256, 0, stream>>>(Hn, DAGG, Pcat, 3*DAGG, NNODE, DAGG, 0);
  k_bcast<<<gN(NNODE*2*DAGG), 256, 0, stream>>>(qst, batch, Pcat, 3*DAGG, DAGG, NNODE, 2*DAGG);
  gemm(Pcat, 3*DAGG, b_pl0, RHD, pl0b, nullptr, 0, z0, RHD, NNODE, 3*DAGG, RHD, 5, pp0, 0);
  gemm(z0,   RHD,    b_pl1, RHD, pl1b, nullptr, 0, z1, RHD, NNODE, RHD,    RHD, 5, pp1, 0);
  gemm(z1,   RHD,    b_pl2, 1,   pl2b, nullptr, 0, ybuf, 1, NNODE, RHD,    1,   0, nullptr, 0);
  k_mask<<<gN(NNODE), 256, 0, stream>>>(ybuf, mask, (float*)d_out, NNODE);
}
